// FPS_PointCNN_24584392802804
// MI455X (gfx1250) — matrix operand order repaired and checked by reading, not compiled, not run on hardware
//
#include <hip/hip_runtime.h>

#define B_      4
#define N_      8192
#define P_      2048
#define KNN_K   16
#define C_IN    64
#define C_OUT   128
#define C_MID   32
#define C_DENSE 64
#define C_CAT   96

typedef __attribute__((ext_vector_type(16))) _Float16 v16h;
typedef __attribute__((ext_vector_type(8)))  float    v8f;
typedef __attribute__((ext_vector_type(2)))  _Float16 h2;
typedef __attribute__((ext_vector_type(4)))  _Float16 h4;
typedef __attribute__((ext_vector_type(4)))  unsigned int u32x4;
typedef __attribute__((ext_vector_type(4)))  int i32x4;
typedef __attribute__((ext_vector_type(8)))  int i32x8;

__device__ inline float eluf(float x){ return x > 0.0f ? x : __expf(x) - 1.0f; }

__device__ inline v8f wmma_f16(v16h a, v16h b, v8f c){
  // v_wmma_f32_16x16x32_f16 (codegen-confirmed 8-arg signature)
  return __builtin_amdgcn_wmma_f32_16x16x32_f16(false, a, false, b, (short)0, c, false, false);
}

// ---- fragment loaders -------------------------------------------------------
template<typename F>
__device__ inline v16h load_a_frag(int lane, F src){          // scalar element src(m,k)
  const int m = lane & 15, h = (lane >> 4) & 1;
  v16h a;
#pragma unroll
  for (int v = 0; v < 8; ++v){
    const int kb = (v < 4 ? 2*v : 16 + 2*(v-4)) + 8*h;
    a[2*v]   = src(m, kb);
    a[2*v+1] = src(m, kb+1);
  }
  return a;
}
template<typename F2>
__device__ inline v16h load_a_frag2(int lane, F2 src2){       // paired src2(m,kb) -> h2
  const int m = lane & 15, h = (lane >> 4) & 1;
  v16h a;
#pragma unroll
  for (int v = 0; v < 8; ++v){
    const int kb = (v < 4 ? 2*v : 16 + 2*(v-4)) + 8*h;
    const h2 t = src2(m, kb);
    a[2*v] = t.x; a[2*v+1] = t.y;
  }
  return a;
}
template<typename F>
__device__ inline v16h load_b_frag(int lane, F src){          // scalar element src(k,n)
  const int n = lane & 15, g = (lane >> 4) & 1;
  v16h b;
#pragma unroll
  for (int v = 0; v < 8; ++v){
    const int k0 = (v < 4 ? 2*v : 16 + 2*(v-4)) + 8*g;
    b[2*v]   = src(k0, n);
    b[2*v+1] = src(k0+1, n);
  }
  return b;
}
template<typename F2>
__device__ inline v16h load_b_frag2(int lane, F2 src2){       // paired src2(k0,n) -> h2
  const int n = lane & 15, g = (lane >> 4) & 1;
  v16h b;
#pragma unroll
  for (int v = 0; v < 8; ++v){
    const int k0 = (v < 4 ? 2*v : 16 + 2*(v-4)) + 8*g;
    const h2 t = src2(k0, n);
    b[2*v] = t.x; b[2*v+1] = t.y;
  }
  return b;
}

// ---------------------------------------------------------------------------
// Kernel 1: farthest point sampling, one workgroup per batch; points and
// min-distances register-resident (8 pts/thread).
// ---------------------------------------------------------------------------
__global__ __launch_bounds__(1024) void k_fps(const float* __restrict__ pts,
                                              float* __restrict__ rep){
  __shared__ float rv[32];
  __shared__ int   ri[32];
  __shared__ int   sfar;
  const int b = blockIdx.x;
  const float* p = pts + (size_t)b * N_ * 3;
  const int tid = threadIdx.x, lane = tid & 31, wid = tid >> 5;

  float px[8], py[8], pz[8], dist[8];
#pragma unroll
  for (int r = 0; r < 8; ++r){
    const int i = tid + r*1024;
    const float3 q = *(const float3*)(p + (size_t)i*3);     // global_load_b96
    px[r] = q.x; py[r] = q.y; pz[r] = q.z;
    dist[r] = 1e10f;
  }
  if (tid == 0) sfar = 0;
  __syncthreads();

  for (int it = 0; it < P_; ++it){
    const int far = sfar;                                    // uniform -> scalar loads
    const float cx = p[far*3+0], cy = p[far*3+1], cz = p[far*3+2];
    if (tid == 0){
      rep[((size_t)b*P_ + it)*3 + 0] = cx;
      rep[((size_t)b*P_ + it)*3 + 1] = cy;
      rep[((size_t)b*P_ + it)*3 + 2] = cz;
    }
    float bv = -1.0f; int bi = 0;
#pragma unroll
    for (int r = 0; r < 8; ++r){
      const float dx = px[r]-cx, dy = py[r]-cy, dz = pz[r]-cz;
      const float nd = fminf(dist[r], dx*dx + dy*dy + dz*dz);
      dist[r] = nd;
      if (nd > bv){ bv = nd; bi = tid + r*1024; }
    }
#pragma unroll
    for (int off = 16; off; off >>= 1){
      const float ov = __shfl_down(bv, off, 32);
      const int   oi = __shfl_down(bi, off, 32);
      if (ov > bv || (ov == bv && oi < bi)){ bv = ov; bi = oi; }
    }
    if (lane == 0){ rv[wid] = bv; ri[wid] = bi; }
    __syncthreads();
    if (wid == 0){
      bv = rv[lane]; bi = ri[lane];
#pragma unroll
      for (int off = 16; off; off >>= 1){
        const float ov = __shfl_down(bv, off, 32);
        const int   oi = __shfl_down(bi, off, 32);
        if (ov > bv || (ov == bv && oi < bi)){ bv = ov; bi = oi; }
      }
      if (lane == 0) sfar = bi;
    }
    __syncthreads();
  }
}

// ---------------------------------------------------------------------------
__global__ __launch_bounds__(256) void k_cvt(const float* __restrict__ in,
                                             _Float16* __restrict__ out){
  const int t = (blockIdx.x * 256 + threadIdx.x) * 4;
  const float4 v = *(const float4*)(in + t);
  h4 o; o.x = (_Float16)v.x; o.y = (_Float16)v.y; o.z = (_Float16)v.z; o.w = (_Float16)v.w;
  *(h4*)(out + t) = o;
}

// ---------------------------------------------------------------------------
__global__ __launch_bounds__(256) void k_prepw(const float* __restrict__ dW,
                                               const float* __restrict__ cW,
                                               const float* __restrict__ d1W,
                                               const float* __restrict__ d2W,
                                               _Float16* __restrict__ Wt,
                                               _Float16* __restrict__ cWt,
                                               _Float16* __restrict__ d1t,
                                               _Float16* __restrict__ d2t){
  const int tid = threadIdx.x;
  for (int t = tid; t < 64*64; t += 256){
    const int n = t >> 6, k = t & 63;
    Wt[n*64 + k] = (_Float16)dW[k*64 + n];
  }
  for (int t = tid; t < 256*48; t += 256){
    const int o = t / 48, kk = t - o*48;
    const int kw = kk / 3, c = kk - 3*kw;
    cWt[o*48 + kk] = (_Float16)cW[o*48 + c*16 + kw];
  }
  for (int t = tid; t < 32*32; t += 256){
    const int n = t >> 5, k = t & 31;
    d1t[n*32 + k] = (k < 3) ? (_Float16)d1W[k*32 + n] : (_Float16)0.0f;
  }
  for (int t = tid; t < 32*32; t += 256){
    const int n = t >> 5, k = t & 31;
    d2t[n*32 + k] = (_Float16)d2W[k*32 + n];
  }
}

// ---------------------------------------------------------------------------
// Kernel 2: fts_d = elu(fts @ dense_W + b). The 8 KB f16 weight matrix is
// DMA'd into LDS once per block by the Tensor Data Mover (one wave issues a
// 1-D descriptor, tile = 4096 x 1 f16 elements), then all 8 waves read their
// B-fragments from LDS.
// ---------------------------------------------------------------------------
__global__ __launch_bounds__(256) void k_dense(const _Float16* __restrict__ ftsh,
                                               const _Float16* __restrict__ Wt,
                                               const float* __restrict__ bias,
                                               _Float16* __restrict__ out){
  __shared__ __align__(16) _Float16 sW[64*64];    // 8 KiB
  const int lane = threadIdx.x & 31;

  if (threadIdx.x < 32){
    const unsigned long long ga = (unsigned long long)(const void*)Wt;
    const unsigned lds_off = (unsigned)(uintptr_t)(void*)&sW[0];  // LDS aperture: addr[31:0]
    u32x4 g0;
    g0.x = 1u;                                             // count=1 (valid user D#)
    g0.y = lds_off;                                        // lds_addr (bytes)
    g0.z = (unsigned)(ga & 0xFFFFFFFFu);                   // global_addr[31:0]
    g0.w = (unsigned)((ga >> 32) & 0x01FFFFFFu) | 0x80000000u;  // addr[56:32] | type=2
    i32x8 g1;
    g1[0] = 0x00010000;                                    // wg_mask=0, data_size=1 (2B)
    g1[1] = (int)(4096u << 16);                            // tensor_dim0 = 4096 (low16<<16)
    g1[2] = (int)(1u << 16);                               // dim0_hi=0 | tensor_dim1=1
    g1[3] = (int)(4096u << 16);                            // dim1_hi=0 | tile_dim0=4096
    g1[4] = 1;                                             // tile_dim1=1, tile_dim2=0
    g1[5] = 4096;                                          // tensor_dim0_stride low32
    g1[6] = 0;
    g1[7] = 0;
    i32x4 gz = {0,0,0,0};
#if defined(__clang_major__) && (__clang_major__ >= 23)
    i32x8 gz8 = {0,0,0,0,0,0,0,0};
    __builtin_amdgcn_tensor_load_to_lds(g0, g1, gz, gz, gz8, 0);
#else
    __builtin_amdgcn_tensor_load_to_lds(g0, g1, gz, gz, 0);
#endif
    __builtin_amdgcn_s_wait_tensorcnt(0);
  }
  __syncthreads();

  const int wave = blockIdx.x * 8 + (threadIdx.x >> 5);
  const int row0 = wave * 16;
  const v8f z8 = {0.f,0.f,0.f,0.f,0.f,0.f,0.f,0.f};
  v8f acc[4] = {z8, z8, z8, z8};

  for (int kt = 0; kt < 2; ++kt){
    v16h a = load_a_frag2(lane, [&](int m, int kb)->h2{
      return *(const h2*)(ftsh + (size_t)(row0+m)*C_IN + kt*32 + kb);
    });
#pragma unroll
    for (int nt = 0; nt < 4; ++nt){
      v16h bf = load_b_frag2(lane, [&](int k0, int n)->h2{
        return *(const h2*)(sW + (size_t)(nt*16+n)*C_IN + kt*32 + k0);  // ds_load
      });
      acc[nt] = wmma_f16(a, bf, acc[nt]);
    }
  }
  const int h = (lane >> 4) & 1, n = lane & 15;
#pragma unroll
  for (int nt = 0; nt < 4; ++nt){
#pragma unroll
    for (int r = 0; r < 8; ++r){
      const int m = r + 8*h;
      const float val = eluf(acc[nt][r] + bias[nt*16 + n]);
      out[(size_t)(row0+m)*C_DENSE + nt*16 + n] = (_Float16)val;
    }
  }
}

// ---------------------------------------------------------------------------
__global__ __launch_bounds__(256) void k_knn(const float* __restrict__ pts,
                                             const float* __restrict__ rep,
                                             int* __restrict__ nidx,
                                             _Float16* __restrict__ pl){
  const int lane = threadIdx.x & 31;
  const int pid = blockIdx.x * 8 + (threadIdx.x >> 5);
  const int b = pid / P_;
  const float rx = rep[(size_t)pid*3+0];
  const float ry = rep[(size_t)pid*3+1];
  const float rz = rep[(size_t)pid*3+2];
  const float* p = pts + (size_t)b * N_ * 3;

  float d[KNN_K]; int id[KNN_K];
#pragma unroll
  for (int j = 0; j < KNN_K; ++j){ d[j] = 3e38f; id[j] = 0; }

  for (int i = lane; i < N_; i += 32){
    if (i + 32 < N_) __builtin_prefetch(p + (size_t)(i+32)*3, 0, 0);  // global_prefetch_b8
    const float3 q = *(const float3*)(p + (size_t)i*3);               // global_load_b96
    const float dx = q.x-rx, dy = q.y-ry, dz = q.z-rz;
    float cd = dx*dx + dy*dy + dz*dz;
    if (cd < d[KNN_K-1]){
      int ci = i;
#pragma unroll
      for (int j = 0; j < KNN_K; ++j){
        if (cd < d[j]){
          const float td = d[j]; d[j] = cd; cd = td;
          const int   ti = id[j]; id[j] = ci; ci = ti;
        }
      }
    }
  }
  for (int k = 0; k < KNN_K; ++k){
    float v = d[0]; int wl = lane;
#pragma unroll
    for (int off = 16; off; off >>= 1){
      const float ov = __shfl_xor(v, off, 32);
      const int   ol = __shfl_xor(wl, off, 32);
      if (ov < v || (ov == v && ol < wl)){ v = ov; wl = ol; }
    }
    const int nid = __shfl(id[0], wl, 32);
    if (lane == 0) nidx[(size_t)pid*KNN_K + k] = nid;
    if (lane < 3){
      const float rc = (lane == 0) ? rx : ((lane == 1) ? ry : rz);
      pl[(size_t)pid*48 + k*3 + lane] = (_Float16)(p[nid*3 + lane] - rc);
    }
    if (lane == wl){
#pragma unroll
      for (int j = 0; j < KNN_K-1; ++j){ d[j] = d[j+1]; id[j] = id[j+1]; }
      d[KNN_K-1] = 3e38f;
    }
  }
}

// ---------------------------------------------------------------------------
__global__ __launch_bounds__(256) void k_convx(const _Float16* __restrict__ pl,
                                               const _Float16* __restrict__ cWt,
                                               const float* __restrict__ convB,
                                               float* __restrict__ X0){
  const int lane = threadIdx.x & 31;
  const int wave = blockIdx.x * 8 + (threadIdx.x >> 5);
  const int mt = wave >> 4, nt = wave & 15;
  const h2 z2 = {(_Float16)0.0f, (_Float16)0.0f};
  v8f acc = {0.f,0.f,0.f,0.f,0.f,0.f,0.f,0.f};

  for (int kt = 0; kt < 2; ++kt){
    v16h a = load_a_frag2(lane, [&](int m, int kb)->h2{
      const int kk = kt*32 + kb;
      return (kk < 48) ? *(const h2*)(pl + (size_t)(mt*16+m)*48 + kk) : z2;
    });
    v16h bf = load_b_frag2(lane, [&](int k0, int n)->h2{
      const int kk = kt*32 + k0;
      return (kk < 48) ? *(const h2*)(cWt + (size_t)(nt*16+n)*48 + kk) : z2;
    });
    acc = wmma_f16(a, bf, acc);
  }
  const int h = (lane >> 4) & 1, n = lane & 15, o = nt*16 + n;
#pragma unroll
  for (int r = 0; r < 8; ++r){
    const int m = r + 8*h;
    X0[(size_t)(mt*16+m)*256 + o] = eluf(acc[r] + convB[o]);
  }
}

// ---------------------------------------------------------------------------
// Kernel 5: per-point tail. One wave per point, 4 waves per block.
// X0 / X1 staged transposed in LDS so the depthwise reductions read float4.
// ---------------------------------------------------------------------------
__global__ __launch_bounds__(128) void k_tail(const int* __restrict__ nidx,
                                              const _Float16* __restrict__ plh,
                                              const _Float16* __restrict__ ftsd,
                                              const float* __restrict__ X0g,
                                              const _Float16* __restrict__ d1t, const float* __restrict__ d1b,
                                              const _Float16* __restrict__ d2t, const float* __restrict__ d2b,
                                              const float* __restrict__ dw1W, const float* __restrict__ dw1b,
                                              const float* __restrict__ dw2W, const float* __restrict__ dw2b,
                                              const float* __restrict__ sdwW, const float* __restrict__ sdwb,
                                              const float* __restrict__ spwW, const float* __restrict__ spwb,
                                              float* __restrict__ outp){
  __shared__ __align__(16) float    sAT[4][256];     // X0^T
  __shared__ __align__(16) float    sBT[4][256];     // X1^T
  __shared__ __align__(16) _Float16 sXh[4][256];     // X2 (row-major, f16)
  __shared__ __align__(16) _Float16 sCat[4][16*96];
  __shared__ __align__(16) float    sFX[4][16*96];
  __shared__ __align__(16) float    sD[4][192];
  const int lane = threadIdx.x & 31, wid = threadIdx.x >> 5;
  const int pid = blockIdx.x * 4 + wid;
  const int b = pid / P_;
  const v8f z8 = {0.f,0.f,0.f,0.f,0.f,0.f,0.f,0.f};
  const h2  z2 = {(_Float16)0.0f, (_Float16)0.0f};

  // 1: load X0 tile, transposed into LDS: sAT[j*16+i] = X0[i,j]
  for (int t = lane; t < 256; t += 32)
    sAT[wid][(t & 15)*16 + (t >> 4)] = X0g[(size_t)pid*256 + t];
  __syncthreads();

  // 2: dw1: X1[j,m] = elu(sum_i X0[i,j]*W[j,m,i] + b); store transposed
#pragma unroll
  for (int e = 0; e < 8; ++e){
    const int t = lane*8 + e, j = t >> 4, m = t & 15;
    float acc = dw1b[j*16 + m];
#pragma unroll
    for (int i4 = 0; i4 < 4; ++i4){
      const float4 x = *(const float4*)(&sAT[wid][j*16 + i4*4]);
      const float4 w = *(const float4*)(dw1W + j*256 + m*16 + i4*4);
      acc += x.x*w.x + x.y*w.y + x.z*w.z + x.w*w.w;
    }
    sBT[wid][m*16 + j] = eluf(acc);                  // X1^T
  }
  __syncthreads();

  // 3: dw2 (no elu): X2[j,m] = sum_i X1[i,j]*W[j,m,i] + b; row-major f16
#pragma unroll
  for (int e = 0; e < 8; ++e){
    const int t = lane*8 + e, j = t >> 4, m = t & 15;
    float acc = dw2b[j*16 + m];
#pragma unroll
    for (int i4 = 0; i4 < 4; ++i4){
      const float4 x = *(const float4*)(&sBT[wid][j*16 + i4*4]);
      const float4 w = *(const float4*)(dw2W + j*256 + m*16 + i4*4);
      acc += x.x*w.x + x.y*w.y + x.z*w.z + x.w*w.w;
    }
    sXh[wid][j*16 + m] = (_Float16)acc;
  }
  __syncthreads();

  // 4: lift1 = elu(pl @ d1_W + b), WMMA (K padded 3->32) -> sCat cols 0..31
  {
    v16h a = load_a_frag(lane, [&](int m, int k)->_Float16{
      return k < 3 ? plh[(size_t)pid*48 + m*3 + k] : (_Float16)0.0f;
    });
    const int h = (lane >> 4) & 1, n = lane & 15;
#pragma unroll
    for (int nt = 0; nt < 2; ++nt){
      v16h bf = load_b_frag2(lane, [&](int k0, int nn)->h2{
        return *(const h2*)(d1t + (size_t)(nt*16+nn)*32 + k0);
      });
      v8f c = wmma_f16(a, bf, z8);
#pragma unroll
      for (int r = 0; r < 8; ++r){
        const int m = r + 8*h;
        sCat[wid][m*96 + nt*16 + n] = (_Float16)eluf(c[r] + d1b[nt*16 + n]);
      }
    }
  }
  __syncthreads();

  // 5: lift2 = elu(lift1 @ d2_W + b), overwrite sCat cols 0..31
  {
    v16h a = load_a_frag2(lane, [&](int m, int kb)->h2{
      return *(const h2*)(&sCat[wid][m*96 + kb]);
    });
    v8f c[2];
#pragma unroll
    for (int nt = 0; nt < 2; ++nt){
      v16h bf = load_b_frag2(lane, [&](int k0, int nn)->h2{
        return *(const h2*)(d2t + (size_t)(nt*16+nn)*32 + k0);
      });
      c[nt] = wmma_f16(a, bf, z8);
    }
    __syncthreads();
    const int h = (lane >> 4) & 1, n = lane & 15;
#pragma unroll
    for (int nt = 0; nt < 2; ++nt){
#pragma unroll
      for (int r = 0; r < 8; ++r){
        const int m = r + 8*h;
        sCat[wid][m*96 + nt*16 + n] = (_Float16)eluf(c[nt][r] + d2b[nt*16 + n]);
      }
    }
  }
  // 6: gather fts_reg into sCat cols 32..95 (paired 32-bit moves)
  for (int t = lane; t < 512; t += 32){
    const int k = t >> 5, cc = (t & 31) * 2;
    const int nid = nidx[(size_t)pid*16 + k];
    *(h2*)(&sCat[wid][k*96 + 32 + cc]) = *(const h2*)(ftsd + ((size_t)b*N_ + nid)*64 + cc);
  }
  __syncthreads();

  // 7: fts_X = X2 @ fts_cat  (16x16 x 16x96, K padded to 32), WMMA
  {
    v16h a = load_a_frag2(lane, [&](int m, int kb)->h2{
      return (kb < 16) ? *(const h2*)(&sXh[wid][m*16 + kb]) : z2;
    });
    const int h = (lane >> 4) & 1, n = lane & 15;
#pragma unroll
    for (int nt = 0; nt < 6; ++nt){
      v16h bf = load_b_frag(lane, [&](int k, int nn)->_Float16{
        return k < 16 ? sCat[wid][k*96 + nt*16 + nn] : (_Float16)0.0f;
      });
      v8f c = wmma_f16(a, bf, z8);
#pragma unroll
      for (int r = 0; r < 8; ++r){
        const int m = r + 8*h;
        sFX[wid][m*96 + nt*16 + n] = c[r];
      }
    }
  }
  __syncthreads();

  // 8: separable depthwise: d[c,m] = sum_k fts_X[k,c]*W[c,m,k] + b[c,m]
#pragma unroll
  for (int e = 0; e < 6; ++e){
    const int t = lane*6 + e, c = t >> 1, m = t & 1;
    float acc = sdwb[c*2 + m];
#pragma unroll
    for (int k = 0; k < 16; ++k) acc += sFX[wid][k*96 + c] * sdwW[c*32 + m*16 + k];
    sD[wid][t] = acc;
  }
  __syncthreads();

  // 9: pointwise: out[o] = elu(sum_{c,m} d[c,m]*W[o,c,m] + b[o]) — float4 both sides
#pragma unroll
  for (int e = 0; e < 4; ++e){
    const int o = lane*4 + e;
    float acc = spwb[o];
#pragma unroll
    for (int t4 = 0; t4 < 48; ++t4){
      const float4 dv = *(const float4*)(&sD[wid][t4*4]);
      const float4 wv = *(const float4*)(spwW + (size_t)o*192 + t4*4);
      acc += dv.x*wv.x + dv.y*wv.y + dv.z*wv.z + dv.w*wv.w;
    }
    outp[(size_t)pid*C_OUT + o] = eluf(acc);
  }
}

// ---------------------------------------------------------------------------
extern "C" void kernel_launch(void* const* d_in, const int* in_sizes, int n_in,
                              void* d_out, int out_size, void* d_ws, size_t ws_size,
                              hipStream_t stream){
  (void)in_sizes; (void)n_in; (void)out_size; (void)ws_size;
  const float* pts  = (const float*)d_in[0];
  const float* fts  = (const float*)d_in[1];
  const float* dW   = (const float*)d_in[2];
  const float* db   = (const float*)d_in[3];
  const float* d1W  = (const float*)d_in[4];
  const float* d1b  = (const float*)d_in[5];
  const float* d2W  = (const float*)d_in[6];
  const float* d2b  = (const float*)d_in[7];
  const float* cW   = (const float*)d_in[8];
  const float* cb   = (const float*)d_in[9];
  const float* dw1W = (const float*)d_in[10];
  const float* dw1b = (const float*)d_in[11];
  const float* dw2W = (const float*)d_in[12];
  const float* dw2b = (const float*)d_in[13];
  const float* sdwW = (const float*)d_in[14];
  const float* sdwb = (const float*)d_in[15];
  const float* spwW = (const float*)d_in[16];
  const float* spwb = (const float*)d_in[17];

  float* rep  = (float*)d_out;                       // [B,P,3]
  float* outp = (float*)d_out + (size_t)B_*P_*3;     // [B,P,128]

  char* ws = (char*)d_ws;
  _Float16* ftsh = (_Float16*)ws;                        // fts in f16      (4 MiB)
  _Float16* ftsd = (_Float16*)(ws + 4194304);            // elu(dense) f16  (4 MiB)
  int*      nidx = (int*)(ws + 8388608);                 // B*P*16 int      (512 KiB)
  _Float16* pl   = (_Float16*)(ws + 8912896);            // B*P*48 f16      (768 KiB)
  float*    X0   = (float*)(ws + 9699328);               // B*P*256 f32     (8 MiB)
  _Float16* Wt   = (_Float16*)(ws + 18087936);           // 64x64 f16
  _Float16* cWt  = (_Float16*)(ws + 18096128);           // 256x48 f16
  _Float16* d1t  = (_Float16*)(ws + 18120704);           // 32x32 f16 (padded)
  _Float16* d2t  = (_Float16*)(ws + 18122752);           // 32x32 f16

  k_fps  <<<B_, 1024, 0, stream>>>(pts, rep);
  k_cvt  <<<(B_*N_*C_IN)/(256*4), 256, 0, stream>>>(fts, ftsh);
  k_prepw<<<1, 256, 0, stream>>>(dW, cW, d1W, d2W, Wt, cWt, d1t, d2t);
  k_dense<<<(B_*N_/16)/8, 256, 0, stream>>>(ftsh, Wt, db, ftsd);
  k_knn  <<<(B_*P_)/8, 256, 0, stream>>>(pts, rep, nidx, pl);
  k_convx<<<((B_*P_/16)*16)/8, 256, 0, stream>>>(pl, cWt, cb, X0);
  k_tail <<<(B_*P_)/4, 128, 0, stream>>>(nidx, pl, ftsd, X0,
                                         d1t, d1b, d2t, d2b,
                                         dw1W, dw1b, dw2W, dw2b,
                                         sdwW, sdwb, spwW, spwb, outp);
}
